// GCNLSTM_83623013253628
// MI455X (gfx1250) — matrix operand order repaired and checked by reading, not compiled, not run on hardware
//
#include <hip/hip_runtime.h>
#include <hip/hip_bf16.h>
#include <math.h>

typedef __attribute__((ext_vector_type(16))) _Float16 v16h;
typedef __attribute__((ext_vector_type(8)))  float    v8f;

#define NEG_SLOPE 0.1f
#define BN_EPS    1e-5f

static const int Bsz = 128;
static const int Ssz = 256;
static const int Fin = 64;
static const int Hsz = 256;
static const int Nn  = Bsz * Ssz;      // 32768 nodes
static const int Esz = 524288;

// ---------------------------------------------------------------------------
// WMMA fragment helpers (CDNA5 wave32, V_WMMA_F32_16X16X32_F16 layouts)
// A (16xK=32, f16): lane 0-15 row=lane, elems 0..7 -> K 0..7, 8..15 -> K 16..23
//                   lane16-31 row=lane-16, elems 0..7 -> K 8..15, 8..15 -> K 24..31
// B (K=32x16, f16): lanes 0-15 col=lane,   elems 0..7 -> K 0..7, 8..15 -> K 16..23
//                   lanes16-31 col=lane-16, elems 0..7 -> K 8..15, 8..15 -> K 24..31
// C/D (16x16 f32): VGPR i -> row = (lane>=16 ? 8 : 0)+i, col = lane&15
// ---------------------------------------------------------------------------
__device__ __forceinline__ v16h load_a_frag(const _Float16* __restrict__ A,
                                            int lda, int lane) {
  int h = lane >> 4, r = lane & 15;
  const _Float16* p0 = A + (size_t)r * lda + h * 8;
  const _Float16* p1 = p0 + 16;
  v16h a;
#pragma unroll
  for (int e = 0; e < 8; ++e) { a[e] = p0[e]; a[8 + e] = p1[e]; }
  return a;
}

// Bt is [Nout, K] row-major f16 ("weights as [out,in]"): B[k,n] = Bt[n*K+k]
__device__ __forceinline__ v16h load_b_frag(const _Float16* __restrict__ Bt,
                                            int K, int lane) {
  int h = lane >> 4, c = lane & 15;
  const _Float16* p0 = Bt + (size_t)c * K + h * 8;
  const _Float16* p1 = p0 + 16;
  v16h b;
#pragma unroll
  for (int e = 0; e < 8; ++e) { b[e] = p0[e]; b[8 + e] = p1[e]; }
  return b;
}

__device__ __forceinline__ v8f wmma_acc(v16h a, v16h b, v8f c) {
  return __builtin_amdgcn_wmma_f32_16x16x32_f16(false, a, false, b,
                                                (short)0, c, false, false);
}

__device__ __forceinline__ float sigmoidf_(float x) { return 1.0f / (1.0f + expf(-x)); }
__device__ __forceinline__ float leakyf_(float x)   { return x >= 0.0f ? x : NEG_SLOPE * x; }

// ---------------------------------------------------------------------------
// Generic tiled WMMA GEMM: C[M,N] = A[M,K] (f16) * Bt[N,K]^T (f16), f32 accum
// one 16x16 tile per wave
// ---------------------------------------------------------------------------
__global__ void gemm_wmma_kernel(const _Float16* __restrict__ A,
                                 const _Float16* __restrict__ Bt,
                                 float* __restrict__ C,
                                 int M, int Ncol, int K) {
  int wave = (blockIdx.x * blockDim.x + threadIdx.x) >> 5;
  int lane = threadIdx.x & 31;
  int ntn = Ncol >> 4;
  int tiles = (M >> 4) * ntn;
  if (wave >= tiles) return;                 // wave-uniform exit
  int m0 = (wave / ntn) << 4;
  int n0 = (wave % ntn) << 4;
  v8f acc = {};
  for (int k0 = 0; k0 < K; k0 += 32) {
    v16h a = load_a_frag(A + (size_t)m0 * K + k0, K, lane);
    v16h b = load_b_frag(Bt + (size_t)n0 * K + k0, K, lane);
    acc = wmma_acc(a, b, acc);
  }
  int h = lane >> 4, c = lane & 15;
#pragma unroll
  for (int i = 0; i < 8; ++i)
    C[(size_t)(m0 + h * 8 + i) * Ncol + n0 + c] = acc[i];
}

// ---------------------------------------------------------------------------
// Fused LSTM step: both directions, gates GEMM (WMMA) + cell update
// block = 128 thr = 4 waves = {i,f,g,o} tiles for one 16x16 h-tile
// grid = (B/16=8, H/16=16, 2 dirs)
// ---------------------------------------------------------------------------
__global__ void lstm_step_kernel(const _Float16* __restrict__ xh,    // [N, H]
                                 const _Float16* __restrict__ WhhT,  // [2][4H][H]
                                 const _Float16* __restrict__ WihT,  // [2][4H][H]
                                 const float*    __restrict__ gbias, // [2][4H]
                                 const _Float16* __restrict__ h_in,  // [2][B][H]
                                 _Float16*       __restrict__ h_out, // [2][B][H]
                                 float*          __restrict__ cst,   // [2][B][H]
                                 float*          __restrict__ lstm_out, // [B][S][2H]
                                 int t) {
  __shared__ float gl[4][16][16];
  int dir  = blockIdx.z;
  int s    = dir ? (Ssz - 1 - t) : t;
  int gate = threadIdx.x >> 5;
  int lane = threadIdx.x & 31;
  int m0 = blockIdx.x << 4;
  int j0 = blockIdx.y << 4;
  int n0 = gate * Hsz + j0;

  const _Float16* Wh = WhhT + ((size_t)dir * 4 * Hsz + n0) * Hsz;
  const _Float16* Wi = WihT + ((size_t)dir * 4 * Hsz + n0) * Hsz;
  const _Float16* hp = h_in + ((size_t)dir * Bsz + m0) * Hsz;
  const _Float16* xp = xh + ((size_t)m0 * Ssz + s) * Hsz;  // row stride S*H

  v8f acc = {};
#pragma unroll
  for (int k0 = 0; k0 < Hsz; k0 += 32) {            // h_prev @ Whh^T
    v16h a = load_a_frag(hp + k0, Hsz, lane);
    v16h b = load_b_frag(Wh + k0, Hsz, lane);
    acc = wmma_acc(a, b, acc);
  }
#pragma unroll
  for (int k0 = 0; k0 < Hsz; k0 += 32) {            // x_t @ Wih^T (fused)
    v16h a = load_a_frag(xp + k0, Ssz * Hsz, lane);
    v16h b = load_b_frag(Wi + k0, Hsz, lane);
    acc = wmma_acc(a, b, acc);
  }
  int hh = lane >> 4, cc = lane & 15;
#pragma unroll
  for (int i = 0; i < 8; ++i) gl[gate][hh * 8 + i][cc] = acc[i];
  __syncthreads();

  const float* bb = gbias + (size_t)dir * 4 * Hsz;
  for (int idx = threadIdx.x; idx < 256; idx += 128) {
    int m = idx >> 4, j = idx & 15;
    int col = j0 + j;
    float gi = gl[0][m][j] + bb[0 * Hsz + col];
    float gf = gl[1][m][j] + bb[1 * Hsz + col];
    float gg = gl[2][m][j] + bb[2 * Hsz + col];
    float go = gl[3][m][j] + bb[3 * Hsz + col];
    size_t ci = ((size_t)dir * Bsz + m0 + m) * Hsz + col;
    float cn = sigmoidf_(gf) * cst[ci] + sigmoidf_(gi) * tanhf(gg);
    cst[ci] = cn;
    float hn = sigmoidf_(go) * tanhf(cn);
    h_out[ci] = (_Float16)hn;
    lstm_out[((size_t)(m0 + m) * Ssz + s) * (2 * Hsz) + dir * Hsz + col] = hn;
  }
}

// ---------------------------------------------------------------------------
// GCN support kernels
// ---------------------------------------------------------------------------
__global__ void deg_init_kernel(float* deg) {
  int i = blockIdx.x * blockDim.x + threadIdx.x;
  if (i < Nn) deg[i] = 1.0f;                    // self-loop
}
__global__ void deg_acc_kernel(const int* __restrict__ ei, float* deg) {
  int e = blockIdx.x * blockDim.x + threadIdx.x;
  if (e < Esz) atomicAdd(&deg[ei[Esz + e]], 1.0f);
}
__global__ void dinv_kernel(float* deg) {
  int i = blockIdx.x * blockDim.x + threadIdx.x;
  if (i < Nn) { float d = deg[i]; deg[i] = d > 0.0f ? rsqrtf(d) : 0.0f; }
}
__global__ void scatter_init_kernel(const float* __restrict__ H,
                                    const float* __restrict__ dinv,
                                    const float* __restrict__ bias,
                                    float* __restrict__ X) {
  int idx = blockIdx.x * blockDim.x + threadIdx.x;
  if (idx < Nn * Hsz) {
    int n = idx >> 8, f = idx & 255;
    float dv = dinv[n];
    X[idx] = dv * dv * H[idx] + bias[f];        // self-loop term + bias
  }
}
__global__ void scatter_edge_kernel(const int* __restrict__ ei,
                                    const float* __restrict__ H,
                                    const float* __restrict__ dinv,
                                    float* __restrict__ X) {
  int idx = blockIdx.x * blockDim.x + threadIdx.x;   // E*64 threads, 4 feats each
  if (idx >= Esz * 64) return;
  int e = idx >> 6, f0 = (idx & 63) << 2;
  int s = ei[e], d = ei[Esz + e];
  float norm = dinv[s] * dinv[d];
  const float* hs = H + (size_t)s * Hsz + f0;
  float* xd = X + (size_t)d * Hsz + f0;
  atomicAdd(&xd[0], norm * hs[0]);
  atomicAdd(&xd[1], norm * hs[1]);
  atomicAdd(&xd[2], norm * hs[2]);
  atomicAdd(&xd[3], norm * hs[3]);
}
__global__ void bn_zero_kernel(float* sums) {
  int i = blockIdx.x * blockDim.x + threadIdx.x;
  if (i < 512) sums[i] = 0.0f;
}
__global__ void bn_stats_kernel(const float* __restrict__ X, float* sums) {
  int f = threadIdx.x;                              // 256 thr, 256 blocks
  float s = 0.0f, s2 = 0.0f;
  for (int r = blockIdx.x; r < Nn; r += gridDim.x) {
    float v = X[(size_t)r * Hsz + f];
    s += v; s2 += v * v;
  }
  atomicAdd(&sums[f], s);
  atomicAdd(&sums[256 + f], s2);
}
__global__ void bn_apply_kernel(float* __restrict__ X, _Float16* __restrict__ xh,
                                const float* __restrict__ sums,
                                const float* __restrict__ gamma,
                                const float* __restrict__ beta) {
  int idx = blockIdx.x * blockDim.x + threadIdx.x;
  if (idx < Nn * Hsz) {
    int f = idx & 255;
    float mu  = sums[f] * (1.0f / Nn);
    float var = sums[256 + f] * (1.0f / Nn) - mu * mu;
    float v = gamma[f] * (X[idx] - mu) * rsqrtf(var + BN_EPS) + beta[f];
    v = leakyf_(v);
    X[idx]  = v;
    xh[idx] = (_Float16)v;
  }
}

// ---------------------------------------------------------------------------
// conversions / zero / bias
// ---------------------------------------------------------------------------
__global__ void cvt_f16_kernel(const float* __restrict__ a, _Float16* __restrict__ b, int n) {
  int i = blockIdx.x * blockDim.x + threadIdx.x;
  if (i < n) b[i] = (_Float16)a[i];
}
__global__ void cvt_trans_f16_kernel(const float* __restrict__ W, _Float16* __restrict__ Wt,
                                     int K, int Ncol) {   // W[K,N] -> Wt[N,K]
  int idx = blockIdx.x * blockDim.x + threadIdx.x;
  if (idx < K * Ncol) {
    int k = idx / Ncol, n = idx % Ncol;
    Wt[(size_t)n * K + k] = (_Float16)W[idx];
  }
}
__global__ void bias_comb_kernel(const float* a, const float* b, float* o, int n) {
  int i = blockIdx.x * blockDim.x + threadIdx.x;
  if (i < n) o[i] = a[i] + b[i];
}
__global__ void zero_f32_kernel(float* p, int n) {
  int i = blockIdx.x * blockDim.x + threadIdx.x;
  if (i < n) p[i] = 0.0f;
}
__global__ void zero_f16_kernel(_Float16* p, int n) {
  int i = blockIdx.x * blockDim.x + threadIdx.x;
  if (i < n) p[i] = (_Float16)0.0f;
}

// ---------------------------------------------------------------------------
// attention + pool + MLP head
// ---------------------------------------------------------------------------
__global__ void attn_kernel(const float* __restrict__ lstm_out,
                            const float* __restrict__ Wa, const float* __restrict__ ba,
                            float* __restrict__ hcat) {
  __shared__ float sc[256];
  __shared__ float red[256];
  int b = blockIdx.x, tid = threadIdx.x;
  const float* Lb = lstm_out + (size_t)b * Ssz * (2 * Hsz);
  const float* row = Lb + (size_t)tid * (2 * Hsz);
  float acc = ba[0];
  for (int k = 0; k < 2 * Hsz; ++k) acc += row[k] * Wa[k];
  red[tid] = acc;
  __syncthreads();
  for (int off = 128; off > 0; off >>= 1) {
    if (tid < off) red[tid] = fmaxf(red[tid], red[tid + off]);
    __syncthreads();
  }
  float mx = red[0];
  __syncthreads();
  float ex = expf(acc - mx);
  sc[tid] = ex; red[tid] = ex;
  __syncthreads();
  for (int off = 128; off > 0; off >>= 1) {
    if (tid < off) red[tid] += red[tid + off];
    __syncthreads();
  }
  float inv = 1.0f / red[0];
  __syncthreads();
  for (int c0 = tid; c0 < 2 * Hsz; c0 += 256) {
    float s = 0.0f;
    for (int s2 = 0; s2 < Ssz; ++s2) s += sc[s2] * Lb[(size_t)s2 * (2 * Hsz) + c0];
    hcat[(size_t)b * 768 + c0] = s * inv;
  }
}
__global__ void pool_kernel(const float* __restrict__ X, const float* __restrict__ Wp,
                            const float* __restrict__ bp, float* __restrict__ hcat) {
  __shared__ float mean[256];
  int b = blockIdx.x, tid = threadIdx.x;
  const float* Xb = X + (size_t)b * Ssz * Hsz;
  float s = 0.0f;
  for (int r = 0; r < Ssz; ++r) s += Xb[(size_t)r * Hsz + tid];
  mean[tid] = s * (1.0f / Ssz);
  __syncthreads();
  float acc = bp[tid];
  for (int k = 0; k < Hsz; ++k) acc += mean[k] * Wp[(size_t)k * Hsz + tid];
  hcat[(size_t)b * 768 + 2 * Hsz + tid] = acc;
}
__global__ void mlp1_kernel(const float* __restrict__ hcat, const float* __restrict__ W1,
                            const float* __restrict__ b1, float* __restrict__ h1) {
  __shared__ float in[768];
  int b = blockIdx.x, tid = threadIdx.x;           // 256 thr
  for (int i = tid; i < 768; i += 256) in[i] = hcat[(size_t)b * 768 + i];
  __syncthreads();
  float acc = b1[tid];
  for (int k = 0; k < 768; ++k) acc += in[k] * W1[(size_t)k * 256 + tid];
  h1[(size_t)b * 256 + tid] = leakyf_(acc);
}
__global__ void mlp2_kernel(const float* __restrict__ h1, const float* __restrict__ W2,
                            const float* __restrict__ b2, float* __restrict__ h2) {
  __shared__ float in[256];
  int b = blockIdx.x, tid = threadIdx.x;           // 128 thr
  for (int i = tid; i < 256; i += 128) in[i] = h1[(size_t)b * 256 + i];
  __syncthreads();
  float acc = b2[tid];
  for (int k = 0; k < 256; ++k) acc += in[k] * W2[(size_t)k * 128 + tid];
  h2[(size_t)b * 128 + tid] = leakyf_(acc);
}
__global__ void mlp3_kernel(const float* __restrict__ h2, const float* __restrict__ W3,
                            const float* __restrict__ b3, const float* __restrict__ ob,
                            float* __restrict__ out) {
  __shared__ float red[128];
  int b = blockIdx.x, tid = threadIdx.x;           // 128 thr
  red[tid] = h2[(size_t)b * 128 + tid] * W3[tid];
  __syncthreads();
  for (int off = 64; off > 0; off >>= 1) {
    if (tid < off) red[tid] += red[tid + off];
    __syncthreads();
  }
  if (tid == 0) out[b] = red[0] + b3[0] + ob[0];
}

// ---------------------------------------------------------------------------
// host launch
// ---------------------------------------------------------------------------
extern "C" void kernel_launch(void* const* d_in, const int* in_sizes, int n_in,
                              void* d_out, int out_size, void* d_ws, size_t ws_size,
                              hipStream_t stream) {
  const float* x      = (const float*)d_in[0];
  const int*   ei     = (const int*)d_in[1];
  const float* Wg[3]  = {(const float*)d_in[3], (const float*)d_in[7],  (const float*)d_in[11]};
  const float* bg[3]  = {(const float*)d_in[4], (const float*)d_in[8],  (const float*)d_in[12]};
  const float* gam[3] = {(const float*)d_in[5], (const float*)d_in[9],  (const float*)d_in[13]};
  const float* bet[3] = {(const float*)d_in[6], (const float*)d_in[10], (const float*)d_in[14]};
  const float* Wih_f = (const float*)d_in[15]; const float* Whh_f = (const float*)d_in[16];
  const float* bih_f = (const float*)d_in[17]; const float* bhh_f = (const float*)d_in[18];
  const float* Wih_b = (const float*)d_in[19]; const float* Whh_b = (const float*)d_in[20];
  const float* bih_b = (const float*)d_in[21]; const float* bhh_b = (const float*)d_in[22];
  const float* Wa = (const float*)d_in[23]; const float* ba = (const float*)d_in[24];
  const float* Wp = (const float*)d_in[25]; const float* bp = (const float*)d_in[26];
  const float* W1 = (const float*)d_in[27]; const float* b1 = (const float*)d_in[28];
  const float* W2 = (const float*)d_in[29]; const float* b2 = (const float*)d_in[30];
  const float* W3 = (const float*)d_in[31]; const float* b3 = (const float*)d_in[32];
  const float* ob = (const float*)d_in[33];

  // workspace carve-up
  char* base = (char*)d_ws;
  size_t off = 0;
  auto carve = [&](size_t bytes) -> char* {
    char* p = base + off;
    off += (bytes + 255) & ~(size_t)255;
    return p;
  };
  _Float16* xh   = (_Float16*)carve((size_t)Nn * Hsz * 2);       // f16 activations
  float*    Hb   = (float*)   carve((size_t)Nn * Hsz * 4);       // x@W
  float*    Xc   = (float*)   carve((size_t)Nn * Hsz * 4);       // GCN layer output
  float*    dinv = (float*)   carve((size_t)Nn * 4);
  float*    sums = (float*)   carve(512 * 4);
  _Float16* WgT  = (_Float16*)carve((size_t)Hsz * Hsz * 2);
  _Float16* WhhT = (_Float16*)carve((size_t)2 * 4 * Hsz * Hsz * 2);
  _Float16* WihT = (_Float16*)carve((size_t)2 * 4 * Hsz * Hsz * 2);
  float*    gbias= (float*)   carve((size_t)2 * 4 * Hsz * 4);
  _Float16* hpp  = (_Float16*)carve((size_t)2 * 2 * Bsz * Hsz * 2); // [pp][dir][B][H]
  float*    cst  = (float*)   carve((size_t)2 * Bsz * Hsz * 4);
  float*    lout = (float*)   carve((size_t)Bsz * Ssz * 2 * Hsz * 4);
  float*    hcat = (float*)   carve((size_t)Bsz * 768 * 4);
  float*    h1b  = (float*)   carve((size_t)Bsz * 256 * 4);
  float*    h2b  = (float*)   carve((size_t)Bsz * 128 * 4);
  (void)ws_size; (void)n_in; (void)in_sizes; (void)out_size;

  const int T = 256;
  auto blk = [](long long n) { return (int)((n + 255) / 256); };

  // degrees -> dinv
  deg_init_kernel<<<blk(Nn), 256, 0, stream>>>(dinv);
  deg_acc_kernel <<<blk(Esz), 256, 0, stream>>>(ei, dinv);
  dinv_kernel    <<<blk(Nn), 256, 0, stream>>>(dinv);

  // 3 GCN layers
  for (int L = 0; L < 3; ++L) {
    int K = (L == 0) ? Fin : Hsz;
    if (L == 0)
      cvt_f16_kernel<<<blk((long long)Nn * Fin), 256, 0, stream>>>(x, xh, Nn * Fin);
    cvt_trans_f16_kernel<<<blk((long long)K * Hsz), 256, 0, stream>>>(Wg[L], WgT, K, Hsz);
    {
      long long tiles = (long long)(Nn / 16) * (Hsz / 16);
      gemm_wmma_kernel<<<blk(tiles * 32), 256, 0, stream>>>(xh, WgT, Hb, Nn, Hsz, K);
    }
    scatter_init_kernel<<<blk((long long)Nn * Hsz), 256, 0, stream>>>(Hb, dinv, bg[L], Xc);
    scatter_edge_kernel<<<blk((long long)Esz * 64), 256, 0, stream>>>(ei, Hb, dinv, Xc);
    bn_zero_kernel <<<2, 256, 0, stream>>>(sums);
    bn_stats_kernel<<<256, 256, 0, stream>>>(Xc, sums);
    bn_apply_kernel<<<blk((long long)Nn * Hsz), 256, 0, stream>>>(Xc, xh, sums, gam[L], bet[L]);
  }

  // LSTM weight prep (f16) + combined biases
  int whn = 4 * Hsz * Hsz;
  cvt_f16_kernel<<<blk(whn), 256, 0, stream>>>(Whh_f, WhhT, whn);
  cvt_f16_kernel<<<blk(whn), 256, 0, stream>>>(Whh_b, WhhT + whn, whn);
  cvt_f16_kernel<<<blk(whn), 256, 0, stream>>>(Wih_f, WihT, whn);
  cvt_f16_kernel<<<blk(whn), 256, 0, stream>>>(Wih_b, WihT + whn, whn);
  bias_comb_kernel<<<blk(4 * Hsz), 256, 0, stream>>>(bih_f, bhh_f, gbias, 4 * Hsz);
  bias_comb_kernel<<<blk(4 * Hsz), 256, 0, stream>>>(bih_b, bhh_b, gbias + 4 * Hsz, 4 * Hsz);

  // init states
  zero_f16_kernel<<<blk(2 * 2 * Bsz * Hsz), 256, 0, stream>>>(hpp, 2 * 2 * Bsz * Hsz);
  zero_f32_kernel<<<blk(2 * Bsz * Hsz), 256, 0, stream>>>(cst, 2 * Bsz * Hsz);

  // recurrence: one fused kernel per step, both directions
  size_t ppStride = (size_t)2 * Bsz * Hsz;   // elements per ping-pong slot
  dim3 lgrid(Bsz / 16, Hsz / 16, 2);
  for (int t = 0; t < T; ++t) {
    lstm_step_kernel<<<lgrid, 128, 0, stream>>>(
        xh, WhhT, WihT, gbias,
        hpp + (size_t)(t & 1) * ppStride,
        hpp + (size_t)((t + 1) & 1) * ppStride,
        cst, lout, t);
  }

  // head
  attn_kernel<<<Bsz, 256, 0, stream>>>(lout, Wa, ba, hcat);
  pool_kernel<<<Bsz, 256, 0, stream>>>(Xc, Wp, bp, hcat);
  mlp1_kernel<<<Bsz, 256, 0, stream>>>(hcat, W1, b1, h1b);
  mlp2_kernel<<<Bsz, 128, 0, stream>>>(h1b, W2, b2, h2b);
  mlp3_kernel<<<Bsz, 128, 0, stream>>>(h2b, W3, b3, ob, (float*)d_out);
}
